// TTTModel_61778809586256
// MI455X (gfx1250) — matrix operand order repaired and checked by reading, not compiled, not run on hardware
//
#include <hip/hip_runtime.h>
#include <hip/hip_bf16.h>

// ---------------------------------------------------------------------------
// Types / helpers for CDNA5 WMMA (wave32)
// ---------------------------------------------------------------------------
typedef __bf16 bf16_t;
typedef __attribute__((ext_vector_type(16))) __bf16 v16bf;
typedef __attribute__((ext_vector_type(8)))  __bf16 v8bf;
typedef __attribute__((ext_vector_type(8)))  float  v8f;
typedef __attribute__((ext_vector_type(4)))  unsigned int v4u;
typedef __attribute__((ext_vector_type(8)))  int v8i;
typedef __attribute__((ext_vector_type(4)))  int v4i;

__device__ __forceinline__ bf16_t f2bf(float f) {
    union { float f; unsigned u; } x; x.f = f;
    unsigned r = x.u + 0x7FFFu + ((x.u >> 16) & 1u);   // round-to-nearest-even
    unsigned short h = (unsigned short)(r >> 16);
    return __builtin_bit_cast(bf16_t, h);
}

__device__ __forceinline__ v16bf cat16(v8bf lo, v8bf hi) {
    return __builtin_shufflevector(lo, hi, 0,1,2,3,4,5,6,7,8,9,10,11,12,13,14,15);
}

__device__ __forceinline__ v8f wmma_bf16(v16bf a, v16bf b, v8f c) {
    return __builtin_amdgcn_wmma_f32_16x16x32_bf16(
        false, a, false, b, (short)0, c, false, false);
}

// A fragment: A row-major [M,K] (ld multiple of 8). Two aligned 16B loads.
// lanes 0-15: M=row, K in {k0..k0+7, k0+16..k0+23}; lanes 16-31: K +8
__device__ __forceinline__ v16bf load_frag_a(const bf16_t* A, int lda,
                                             int row0, int k0, int lane) {
    const bf16_t* p = A + (row0 + (lane & 15)) * lda + k0 + ((lane >> 4) << 3);
    v8bf lo = *(const v8bf*)p;
    v8bf hi = *(const v8bf*)(p + 16);
    return cat16(lo, hi);
}

// B fragment from B^T stored row-major: BT[n][k]. Two aligned 16B loads.
// lanes 0-15: K in {k0..k0+7, k0+16..k0+23}; lanes 16-31: K +8; col = lane&15
__device__ __forceinline__ v16bf load_frag_bT(const bf16_t* BT, int ldb,
                                              int col0, int k0, int lane) {
    const bf16_t* p = BT + (col0 + (lane & 15)) * ldb + k0 + ((lane >> 4) << 3);
    v8bf lo = *(const v8bf*)p;
    v8bf hi = *(const v8bf*)(p + 16);
    return cat16(lo, hi);
}

// ---------------------------------------------------------------------------
// Model constants
// ---------------------------------------------------------------------------
#define SEQ_LEN 1024
#define BATCH   4
#define ROWS    4096
#define D_MODEL 512
#define F_DIM   128
#define MB      16
#define CHUNK_R 64
#define NCHUNK  64
#define VOCAB   32000
#define FF      16384

#define LD_W    136     // padded ld for W in LDS (bank-conflict free, 16B aligned)
#define LD_DT   72      // padded ld for diff^T in LDS
#define LD_AS   264     // padded ld for TDM-staged A tile (256 + 8)

// ---------------------------------------------------------------------------
// Tensor Data Mover: stage a 64-row x 256-col bf16 tile (row stride 512 elems
// in global) into LDS with row padding 264 elems via D# pad fields.
// ---------------------------------------------------------------------------
__device__ __forceinline__ void tdm_stage_64x256(unsigned lds_off, const bf16_t* gsrc) {
#if defined(__HIP_DEVICE_COMPILE__) && __has_builtin(__builtin_amdgcn_tensor_load_to_lds)
    unsigned long long ga = (unsigned long long)(uintptr_t)gsrc;
    v4u g0;
    g0[0] = 1u;                                        // count=1, user mode
    g0[1] = lds_off;                                   // lds_addr
    g0[2] = (unsigned)(ga & 0xFFFFFFFFull);            // global_addr[31:0]
    g0[3] = (unsigned)((ga >> 32) & 0x01FFFFFFull)     // global_addr[56:32]
          | 0x80000000u;                               // type=2 (image)
    v8i g1;
    g1[0] = (int)0x07910000u;  // data_size=1(2B), pad_en, pad_interval=128dw, pad_amt=4dw
    g1[1] = (int)(512u << 16);         // tensor_dim0[15:0]=512
    g1[2] = (int)(64u  << 16);         // tensor_dim0[31:16]=0 | tensor_dim1[15:0]=64
    g1[3] = (int)(256u << 16);         // tensor_dim1[31:16]=0 | tile_dim0=256
    g1[4] = (int)64;                   // tile_dim1=64 | tile_dim2=0
    g1[5] = (int)512;                  // tensor_dim0_stride[31:0]=512
    g1[6] = 0;
    g1[7] = 0;
    v4i gz = {0, 0, 0, 0};
#if __clang_major__ >= 23
    v8i gz8 = {0, 0, 0, 0, 0, 0, 0, 0};
    __builtin_amdgcn_tensor_load_to_lds(g0, g1, gz, gz, gz8, 0);
#else
    __builtin_amdgcn_tensor_load_to_lds(g0, g1, gz, gz, 0);
#endif
    __builtin_amdgcn_s_wait_tensorcnt(0);
#else
    (void)lds_off; (void)gsrc;
#endif
}

// ---------------------------------------------------------------------------
// fp32 -> bf16 converter (plain) and transpose-converter
// ---------------------------------------------------------------------------
__global__ void __launch_bounds__(256) k_cvt(const float* __restrict__ s,
                                             bf16_t* __restrict__ d, int n) {
    for (int i = blockIdx.x * 256 + threadIdx.x; i < n; i += gridDim.x * 256)
        d[i] = f2bf(s[i]);
}

// dst[c*R + r] = bf16(src[r*C + c])   (src is [R,C])
__global__ void __launch_bounds__(256) k_cvt_t(const float* __restrict__ s,
                                               bf16_t* __restrict__ d,
                                               int R, int C) {
    int n = R * C;
    for (int i = blockIdx.x * 256 + threadIdx.x; i < n; i += gridDim.x * 256) {
        int r = i / C, c = i - r * C;
        d[c * R + r] = f2bf(s[i]);
    }
}

// ---------------------------------------------------------------------------
// Embedding + positional encoding
// ---------------------------------------------------------------------------
__global__ void __launch_bounds__(256) k_embed(const int* __restrict__ src,
                                               const float* __restrict__ emb,
                                               float* __restrict__ x,
                                               bf16_t* __restrict__ xb) {
    int row = blockIdx.x;                 // s*BATCH + b
    int s   = row >> 2;
    int tok = src[row];
    const float sqrtd = 22.62741699796952f;
    for (int d = threadIdx.x; d < D_MODEL; d += 256) {
        int   i   = d >> 1;
        float div = __expf(-9.210340371976184f * (float)(2 * i) / (float)D_MODEL);
        float ang = (float)s * div;
        float pe  = (d & 1) ? __cosf(ang) : __sinf(ang);
        float v   = emb[tok * D_MODEL + d] * sqrtd + pe;
        x [row * D_MODEL + d] = v;
        xb[row * D_MODEL + d] = f2bf(v);
    }
}

// ---------------------------------------------------------------------------
// Fused Q/K/V projections. Also emits KbT [128,4096] (packed 16B stores).
// grid.x = ROWS/64; 8 waves: wave -> (m = wave>>1, 4 col tiles)
// ---------------------------------------------------------------------------
__global__ void __launch_bounds__(256) k_qkvt(const bf16_t* __restrict__ xb,
                                              const bf16_t* __restrict__ tkT, // [128,512]
                                              const bf16_t* __restrict__ tqT,
                                              const bf16_t* __restrict__ tvT,
                                              bf16_t* __restrict__ Kb,
                                              bf16_t* __restrict__ KbT,       // [128,4096]
                                              bf16_t* __restrict__ Qb,
                                              float*  __restrict__ Qf,
                                              float*  __restrict__ Tf) {
    int lane = threadIdx.x & 31, wave = threadIdx.x >> 5;
    int row0 = blockIdx.x * 64 + (wave >> 1) * 16;
    int nB   = (wave & 1) * 4;

    v8f aK[4] = {}, aQ[4] = {}, aV[4] = {};
    for (int k0 = 0; k0 < D_MODEL; k0 += 32) {
        v16bf a = load_frag_a(xb, D_MODEL, row0, k0, lane);
#pragma unroll
        for (int t = 0; t < 4; ++t) {
            int c0 = (nB + t) * 16;
            aK[t] = wmma_bf16(a, load_frag_bT(tkT, D_MODEL, c0, k0, lane), aK[t]);
            aQ[t] = wmma_bf16(a, load_frag_bT(tqT, D_MODEL, c0, k0, lane), aQ[t]);
            aV[t] = wmma_bf16(a, load_frag_bT(tvT, D_MODEL, c0, k0, lane), aV[t]);
        }
    }
#pragma unroll
    for (int t = 0; t < 4; ++t) {
        int col = (nB + t) * 16 + (lane & 15);
        v8bf pk;
#pragma unroll
        for (int r = 0; r < 8; ++r) {
            int row = row0 + r + ((lane >> 4) << 3);
            int idx = row * F_DIM + col;
            bf16_t kv = f2bf(aK[t][r]);
            pk[r]   = kv;
            Kb[idx] = kv;
            Qb[idx] = f2bf(aQ[t][r]);
            Qf[idx] = aQ[t][r];
            Tf[idx] = aV[t][r] - aK[t][r];
        }
        // transposed copy, contiguous 16B per lane
        *(v8bf*)(KbT + (size_t)col * ROWS + row0 + ((lane >> 4) << 3)) = pk;
    }
}

// ---------------------------------------------------------------------------
// Fused inner-LR head, TDM-staged A tile in LDS.
// grid = (FF/128, NCHUNK); wave -> one 16-col tile across 4 M-tiles
// ---------------------------------------------------------------------------
__global__ void __launch_bounds__(256) k_lr(const bf16_t* __restrict__ xb,
                                            const bf16_t* __restrict__ WlrB,  // [FF,512]
                                            const float*  __restrict__ blr,
                                            float* __restrict__ lrm) {
    __shared__ alignas(16) bf16_t As[CHUNK_R * LD_AS];   // 33 KB staged A (64x256 +pad)
    int lane = threadIdx.x & 31, wave = threadIdx.x >> 5;
    int c    = blockIdx.y;
    int col0 = blockIdx.x * 128 + wave * 16;
    int rb   = c * CHUNK_R;

    v8f acc[4] = {};
    for (int kh = 0; kh < D_MODEL; kh += 256) {          // two 64x256 halves
        __syncthreads();                                 // LDS reuse safety
        if (wave == 0)
            tdm_stage_64x256((unsigned)(uintptr_t)&As[0],
                             xb + (size_t)rb * D_MODEL + kh);
        __syncthreads();
        for (int k0 = 0; k0 < 256; k0 += 32) {
            v16bf b = load_frag_bT(WlrB, D_MODEL, col0, kh + k0, lane);
#pragma unroll
            for (int m = 0; m < 4; ++m) {
                v16bf a = load_frag_a(&As[0], LD_AS, m * 16, k0, lane);
                acc[m] = wmma_bf16(a, b, acc[m]);
            }
        }
    }
    int col   = col0 + (lane & 15);
    float bia = blr[col];
    float s = 0.f;
#pragma unroll
    for (int m = 0; m < 4; ++m)
#pragma unroll
        for (int r = 0; r < 8; ++r) {
            float v = acc[m][r] + bia;
            s += 0.01f / (1.f + __expf(-v));
        }
    s += __shfl_xor(s, 16, 32);
    if (lane < 16) lrm[c * FF + col] = s * (1.f / 64.f);
}

// ---------------------------------------------------------------------------
// Sequential TTT scan: one persistent workgroup, 64 chunks, W in LDS.
// ---------------------------------------------------------------------------
__global__ void __launch_bounds__(256) k_scan(const bf16_t* __restrict__ Kb,
                                              const bf16_t* __restrict__ KbT, // [128,4096]
                                              const bf16_t* __restrict__ Qb,
                                              const float*  __restrict__ Qf,
                                              const float*  __restrict__ Tf,
                                              const float*  __restrict__ lrm,
                                              const float*  __restrict__ W0,
                                              const float*  __restrict__ b0,
                                              float*  __restrict__ Wg,
                                              bf16_t* __restrict__ zb) {
    __shared__ alignas(16) bf16_t Wb[F_DIM * LD_W];      // 34 KB, bf16 mirror of W
    __shared__ alignas(16) bf16_t diffT[F_DIM * LD_DT];  // 18 KB, diff^T [col][row]
    __shared__ float gbPart[4][F_DIM];
    __shared__ float gbS[F_DIM];
    __shared__ float bS[F_DIM];

    int tid = threadIdx.x, lane = tid & 31, wave = tid >> 5;
    const float scale = 2.0f / (float)(MB * BATCH * F_DIM);

    for (int i = tid; i < F_DIM * F_DIM; i += 256) {
        float w = W0[i];
        Wg[i] = w;
        Wb[(i >> 7) * LD_W + (i & 127)] = f2bf(w);
    }
    if (tid < F_DIM) bS[tid] = b0[tid];
    __syncthreads();

    for (int c = 0; c < NCHUNK; ++c) {
        const int rb = c * CHUNK_R;

        // --- Phase A: diff = (train@W^T + b) - target ; stage diff^T in LDS ---
        {
            int m  = wave >> 1;
            int nB = (wave & 1) * 4;
            v8f acc[4] = {};
            for (int k0 = 0; k0 < F_DIM; k0 += 32) {
                v16bf a = load_frag_a(Kb + rb * F_DIM, F_DIM, m * 16, k0, lane);
#pragma unroll
                for (int t = 0; t < 4; ++t)
                    acc[t] = wmma_bf16(a, load_frag_bT(&Wb[0], LD_W, (nB + t) * 16, k0, lane), acc[t]);
            }
#pragma unroll
            for (int t = 0; t < 4; ++t) {
                int col = (nB + t) * 16 + (lane & 15);
                float colsum = 0.f;
                v8bf pk;
#pragma unroll
                for (int r = 0; r < 8; ++r) {
                    int row  = m * 16 + r + ((lane >> 4) << 3);
                    float df = (acc[t][r] + bS[col]) - Tf[(rb + row) * F_DIM + col];
                    pk[r] = f2bf(df);
                    colsum += df;
                }
                *(v8bf*)(&diffT[col * LD_DT + m * 16 + ((lane >> 4) << 3)]) = pk;
                colsum += __shfl_xor(colsum, 16, 32);
                if (lane < 16) gbPart[m][col] = colsum;
            }
        }
        __syncthreads();
        if (tid < F_DIM)
            gbS[tid] = gbPart[0][tid] + gbPart[1][tid] + gbPart[2][tid] + gbPart[3][tid];
        __syncthreads();

        // --- Phase B: gW = diff^T @ train ; W -= lr * (2/n) * gW ---
        {
            int mi = wave;
            v8f acc[8] = {};
            for (int k0 = 0; k0 < CHUNK_R; k0 += 32) {
                v16bf a = load_frag_a(&diffT[0], LD_DT, mi * 16, k0, lane);
#pragma unroll
                for (int t = 0; t < 8; ++t)
                    acc[t] = wmma_bf16(a, load_frag_bT(KbT + rb, ROWS, t * 16, k0, lane), acc[t]);
            }
#pragma unroll
            for (int t = 0; t < 8; ++t) {
                int col = t * 16 + (lane & 15);
#pragma unroll
                for (int r = 0; r < 8; ++r) {
                    int row = mi * 16 + r + ((lane >> 4) << 3);
                    int idx = row * F_DIM + col;
                    float g  = acc[t][r] * scale;
                    float lr = lrm[c * FF + idx];
                    float wn = Wg[idx] - lr * g;
                    Wg[idx] = wn;
                    Wb[row * LD_W + col] = f2bf(wn);
                }
            }
        }
        __syncthreads();

        // --- bias update: b -= mean_j(lr) * gb ---
        if (tid < F_DIM) {
            float s = 0.f;
            for (int j = 0; j < F_DIM; ++j) s += lrm[c * FF + tid * F_DIM + j];
            bS[tid] -= (s * (1.f / 128.f)) * (gbS[tid] * scale);
        }
        __syncthreads();

        // --- Phase C: z = test@Wn^T + bn + test ---
        {
            int m  = wave >> 1;
            int nB = (wave & 1) * 4;
            v8f acc[4] = {};
            for (int k0 = 0; k0 < F_DIM; k0 += 32) {
                v16bf a = load_frag_a(Qb + rb * F_DIM, F_DIM, m * 16, k0, lane);
#pragma unroll
                for (int t = 0; t < 4; ++t)
                    acc[t] = wmma_bf16(a, load_frag_bT(&Wb[0], LD_W, (nB + t) * 16, k0, lane), acc[t]);
            }
#pragma unroll
            for (int t = 0; t < 4; ++t) {
                int col = (nB + t) * 16 + (lane & 15);
#pragma unroll
                for (int r = 0; r < 8; ++r) {
                    int row = m * 16 + r + ((lane >> 4) << 3);
                    float z = acc[t][r] + bS[col] + Qf[(rb + row) * F_DIM + col];
                    zb[(rb + row) * F_DIM + col] = f2bf(z);
                }
            }
        }
        __syncthreads();
    }
}

// ---------------------------------------------------------------------------
// Output projection: x = z @ to ; refresh xb
// grid = (D_MODEL/128, ROWS/64)
// ---------------------------------------------------------------------------
__global__ void __launch_bounds__(256) k_outproj(const bf16_t* __restrict__ zb,
                                                 const bf16_t* __restrict__ toT, // [512,128]
                                                 float*  __restrict__ x,
                                                 bf16_t* __restrict__ xb) {
    int lane = threadIdx.x & 31, wave = threadIdx.x >> 5;
    int row0 = blockIdx.y * 64 + (wave >> 1) * 16;
    int col0 = blockIdx.x * 128 + (wave & 1) * 64;

    v8f acc[4] = {};
    for (int k0 = 0; k0 < F_DIM; k0 += 32) {
        v16bf a = load_frag_a(zb, F_DIM, row0, k0, lane);
#pragma unroll
        for (int t = 0; t < 4; ++t)
            acc[t] = wmma_bf16(a, load_frag_bT(toT, F_DIM, col0 + t * 16, k0, lane), acc[t]);
    }
#pragma unroll
    for (int t = 0; t < 4; ++t) {
        int col = col0 + t * 16 + (lane & 15);
#pragma unroll
        for (int r = 0; r < 8; ++r) {
            int row = row0 + r + ((lane >> 4) << 3);
            x [row * D_MODEL + col] = acc[t][r];
            xb[row * D_MODEL + col] = f2bf(acc[t][r]);
        }
    }
}

// ---------------------------------------------------------------------------
// LM head: logits = x @ lm_w^T + lm_b (134 GFLOP, 524 MB NT stores)
// grid = (VOCAB/256, ROWS/64); each wave: 64 rows x 32 cols (8 tiles)
// -> 12 b128 loads per 8 WMMAs; NT stores keep lm_w resident in L2
// ---------------------------------------------------------------------------
__global__ void __launch_bounds__(256) k_lmhead(const bf16_t* __restrict__ xb,
                                                const bf16_t* __restrict__ lmB,  // [VOCAB,512]
                                                const float*  __restrict__ lmb,
                                                float* __restrict__ out) {
    int lane = threadIdx.x & 31, wave = threadIdx.x >> 5;
    int row0 = blockIdx.y * 64;
    int col0 = blockIdx.x * 256 + wave * 32;

    v8f acc[4][2] = {};
    for (int k0 = 0; k0 < D_MODEL; k0 += 32) {
        if (k0 + 32 < D_MODEL)
            __builtin_prefetch(lmB + (size_t)(col0 + (lane & 15)) * D_MODEL + k0 + 32, 0, 1);
        v16bf b0f = load_frag_bT(lmB, D_MODEL, col0,      k0, lane);
        v16bf b1f = load_frag_bT(lmB, D_MODEL, col0 + 16, k0, lane);
#pragma unroll
        for (int m = 0; m < 4; ++m) {
            v16bf a = load_frag_a(xb, D_MODEL, row0 + m * 16, k0, lane);
            acc[m][0] = wmma_bf16(a, b0f, acc[m][0]);
            acc[m][1] = wmma_bf16(a, b1f, acc[m][1]);
        }
    }
#pragma unroll
    for (int t = 0; t < 2; ++t) {
        int col = col0 + t * 16 + (lane & 15);
        float b = lmb[col];
#pragma unroll
        for (int m = 0; m < 4; ++m)
#pragma unroll
            for (int r = 0; r < 8; ++r) {
                int row = row0 + m * 16 + r + ((lane >> 4) << 3);
                __builtin_nontemporal_store(acc[m][t][r] + b,
                                            &out[(size_t)row * VOCAB + col]);
            }
    }
}

// ---------------------------------------------------------------------------
// Host driver
// ---------------------------------------------------------------------------
extern "C" void kernel_launch(void* const* d_in, const int* in_sizes, int n_in,
                              void* d_out, int out_size, void* d_ws, size_t ws_size,
                              hipStream_t stream) {
    const int*   src     = (const int*)  d_in[0];
    const float* emb     = (const float*)d_in[1];
    const float* theta_k = (const float*)d_in[2];
    const float* theta_q = (const float*)d_in[3];
    const float* theta_v = (const float*)d_in[4];
    const float* theta_o = (const float*)d_in[5];
    const float* w0      = (const float*)d_in[6];
    const float* b0      = (const float*)d_in[7];
    const float* lr_w    = (const float*)d_in[8];
    const float* lr_b    = (const float*)d_in[9];
    const float* lm_w    = (const float*)d_in[10];
    const float* lm_b    = (const float*)d_in[11];
    float* out = (float*)d_out;

    char* ws = (char*)d_ws;
    auto alloc = [&](size_t bytes) -> void* {
        void* p = (void*)ws;
        ws += (bytes + 255) & ~(size_t)255;
        return p;
    };
    float*  x    = (float*) alloc((size_t)ROWS * D_MODEL * 4);
    bf16_t* xb   = (bf16_t*)alloc((size_t)ROWS * D_MODEL * 2);
    bf16_t* Kb   = (bf16_t*)alloc((size_t)ROWS * F_DIM * 2);
    bf16_t* KbT  = (bf16_t*)alloc((size_t)F_DIM * ROWS * 2);
    bf16_t* Qb   = (bf16_t*)alloc((size_t)ROWS * F_DIM * 2);
    float*  Qf   = (float*) alloc((size_t)ROWS * F_DIM * 4);
    float*  Tf   = (float*) alloc((size_t)ROWS * F_DIM * 4);
    float*  lrm  = (float*) alloc((size_t)NCHUNK * FF * 4);
    bf16_t* WlrB = (bf16_t*)alloc((size_t)FF * D_MODEL * 2);
    bf16_t* tkT  = (bf16_t*)alloc((size_t)D_MODEL * F_DIM * 2);
    bf16_t* tqT  = (bf16_t*)alloc((size_t)D_MODEL * F_DIM * 2);
    bf16_t* tvT  = (bf16_t*)alloc((size_t)D_MODEL * F_DIM * 2);
    bf16_t* toT  = (bf16_t*)alloc((size_t)F_DIM * D_MODEL * 2);
    bf16_t* lmB  = (bf16_t*)alloc((size_t)VOCAB * D_MODEL * 2);
    float*  Wg   = (float*) alloc((size_t)F_DIM * F_DIM * 4);
    bf16_t* zb   = (bf16_t*)alloc((size_t)ROWS * F_DIM * 2);

    k_cvt<<<dim3(8192), dim3(256), 0, stream>>>(lm_w, lmB, VOCAB * D_MODEL);
    k_embed<<<dim3(ROWS), dim3(256), 0, stream>>>(src, emb, x, xb);

    for (int l = 0; l < 2; ++l) {
        const int pW = l * D_MODEL * F_DIM;
        k_cvt_t<<<dim3(256), dim3(256), 0, stream>>>(theta_k + pW, tkT, D_MODEL, F_DIM);
        k_cvt_t<<<dim3(256), dim3(256), 0, stream>>>(theta_q + pW, tqT, D_MODEL, F_DIM);
        k_cvt_t<<<dim3(256), dim3(256), 0, stream>>>(theta_v + pW, tvT, D_MODEL, F_DIM);
        k_cvt_t<<<dim3(256), dim3(256), 0, stream>>>(theta_o + pW, toT, F_DIM, D_MODEL);
        k_cvt<<<dim3(8192), dim3(256), 0, stream>>>(lr_w + (size_t)l * FF * D_MODEL, WlrB, FF * D_MODEL);

        k_qkvt<<<dim3(ROWS / 64), dim3(256), 0, stream>>>(xb, tkT, tqT, tvT,
                                                          Kb, KbT, Qb, Qf, Tf);
        k_lr<<<dim3(FF / 128, NCHUNK), dim3(256), 0, stream>>>(xb, WlrB, lr_b + l * FF, lrm);
        k_scan<<<dim3(1), dim3(256), 0, stream>>>(Kb, KbT, Qb, Qf, Tf, lrm,
                                                  w0 + l * F_DIM * F_DIM, b0 + l * F_DIM, Wg, zb);
        k_outproj<<<dim3(D_MODEL / 128, ROWS / 64), dim3(256), 0, stream>>>(zb, toT, x, xb);
    }

    k_lmhead<<<dim3(VOCAB / 256, ROWS / 64), dim3(256), 0, stream>>>(xb, lmB, lm_b, out);
}
